// PostPatchAdaptiveGraphLearner_54314156425458
// MI455X (gfx1250) — matrix operand order repaired and checked by reading, not compiled, not run on hardware
//
#include <hip/hip_runtime.h>
#include <hip/hip_bf16.h>
#include <math.h>

// ---------------------------------------------------------------------------
// PostPatchAdaptiveGraphLearner — CDNA5 (gfx1250).
// Heavy matmuls: v_wmma_f32_16x16x32_f16 fed by vectorized (b128) LDS loads.
// GNN panel staging: TDM tensor_load_to_lds (+ s_wait_tensorcnt) when available.
// ---------------------------------------------------------------------------

#define B_   16
#define N_   358
#define P_   72
#define D_   96
#define ND_  64
#define H_   4
#define K_   10
#define NH_  4
#define HD_  24          // D_/NH_
#define BN_  (B_*N_)     // 5728
#define NN_  (N_*N_)     // 128164

typedef __attribute__((ext_vector_type(16))) _Float16 v16h;
typedef __attribute__((ext_vector_type(8)))  _Float16 v8h;
typedef __attribute__((ext_vector_type(8)))  float    v8f;

#define WMMA16(a,b,c) __builtin_amdgcn_wmma_f32_16x16x32_f16(false,(a),false,(b),(short)0,(c),false,false)

#if __has_builtin(__builtin_amdgcn_tensor_load_to_lds) && __has_builtin(__builtin_amdgcn_s_wait_tensorcnt)
#define USE_TDM 1
typedef __attribute__((ext_vector_type(4))) unsigned int tdm_v4u;
typedef __attribute__((ext_vector_type(8))) int          tdm_v8i;
typedef __attribute__((ext_vector_type(4))) int          tdm_v4i;
#endif

// ---------------- wave helpers (wave32) ----------------
__device__ __forceinline__ float wred_sum(float v) {
#pragma unroll
  for (int o = 16; o; o >>= 1) v += __shfl_xor(v, o);
  return v;
}
__device__ __forceinline__ float wred_max(float v) {
#pragma unroll
  for (int o = 16; o; o >>= 1) v = fmaxf(v, __shfl_xor(v, o));
  return v;
}
__device__ __forceinline__ float gelu_exact(float x) {
  return 0.5f * x * (1.0f + erff(x * 0.70710678118654752f));
}
__device__ __forceinline__ float sigmoidf_(float x) { return 1.0f / (1.0f + expf(-x)); }

// ---------------- vectorized WMMA tile loaders ----------------
// All call sites guarantee each 8-half run is either fully valid or fully
// invalid, so validity is a per-run select on a 16-byte vector load.
// Speculative over-reads land inside the (padded) LDS region: harmless.

// A (16x32 f16, row-major [m][k], stride in halves, 16B-aligned runs).
// lane: m = mbase+(lane&15); kb = (lane&16)?16:0.
// halves h = k (kbase+kb+h).
__device__ __forceinline__ v16h ldsA_v(const _Float16* p, int stride, int mbase,
                                       int kbase, int kvalid, int lane) {
  int m  = mbase + (lane & 15);
  int kb = (lane & 16) ? 16 : 0;
  const _Float16* r = p + m * stride + kbase + kb;
  v8h z = {};
  v8h lo = (kbase + kb + 8  <= kvalid) ? *(const v8h*)r        : z;
  v8h hi = (kbase + kb + 16 <= kvalid) ? *(const v8h*)(r + 8) : z;
  return __builtin_shufflevector(lo, hi, 0,1,2,3,4,5,6,7,8,9,10,11,12,13,14,15);
}
// B (32x16 f16) with k CONTIGUOUS per n-row: element (k,n) at p[n*rowstride + kbase + k].
// lane: n = nbase+(lane&15); kb = (lane&16)?16:0; halves h = k (kbase+kb+h).
__device__ __forceinline__ v16h ldsBk_v(const _Float16* p, int rowstride, int nbase,
                                        int kbase, int kvalid, int lane) {
  int n  = nbase + (lane & 15);
  int K0 = kbase + ((lane & 16) ? 16 : 0);
  const _Float16* r = p + n * rowstride + K0;
  v8h z = {};
  v8h lo = (K0 + 8  <= kvalid) ? *(const v8h*)r       : z;
  v8h hi = (K0 + 16 <= kvalid) ? *(const v8h*)(r + 8) : z;
  return __builtin_shufflevector(lo, hi, 0,1,2,3,4,5,6,7,8,9,10,11,12,13,14,15);
}
// C/D f32 store: lane covers col n, rows m0..m0+7 (m0 = mbase + (lane<16?0:8)).
__device__ __forceinline__ void stC_f32(float* p, int stride, int mbase, int nbase,
                                        v8f c, float scale, int lane) {
  int n  = nbase + (lane & 15);
  int m0 = mbase + ((lane & 16) ? 8 : 0);
#pragma unroll
  for (int r = 0; r < 8; ++r) p[(m0 + r) * stride + n] = c[r] * scale;
}
__device__ __forceinline__ v8f initC_bias(const float* bias, int nbase, int lane) {
  float bv = bias[nbase + (lane & 15)];
  v8f c;
#pragma unroll
  for (int r = 0; r < 8; ++r) c[r] = bv;
  return c;
}

// ---------------- wave-level top-k(10) + renormalize + write ----------------
__device__ __forceinline__ void topk_renorm_write(float* sr, int n, float* pv, int* pi,
                                                  float* orow, int lane) {
#pragma unroll 1
  for (int it = 0; it < K_; ++it) {
    float bv = -1.0f; int bi = 0x7fffffff;
    for (int j = lane; j < n; j += 32) {
      float v = sr[j];
      if (v > bv) { bv = v; bi = j; }
    }
#pragma unroll
    for (int o = 16; o; o >>= 1) {
      float ov = __shfl_xor(bv, o); int oi = __shfl_xor(bi, o);
      if (ov > bv || (ov == bv && oi < bi)) { bv = ov; bi = oi; }
    }
    if (lane == 0) { pv[it] = bv; pi[it] = bi; sr[bi] = -1.0f; }
    __builtin_amdgcn_wave_barrier();
  }
  float ts = 0.0f;
#pragma unroll
  for (int it = 0; it < K_; ++it) ts += pv[it];
  float inv = 1.0f / (ts + 1e-8f);
  for (int j = lane; j < n; j += 32) orow[j] = 0.0f;
  __builtin_amdgcn_wave_barrier();
  if (lane < K_) orow[pi[lane]] = pv[lane] * inv;
}

// ===========================================================================
// Kernel 1: static adjacencies (relu -> softmax/T -> topk -> renorm)
// ===========================================================================
__global__ void __launch_bounds__(256) static_adj_kernel(
    const float* __restrict__ le1, const float* __restrict__ le2,
    const float* __restrict__ ge1, const float* __restrict__ ge2,
    const float* __restrict__ temp, float* __restrict__ out) {
  __shared__ float sr[8][360];
  __shared__ float pv[8][K_];
  __shared__ int   pi[8][K_];
  int tid = threadIdx.x, wave = tid >> 5, lane = tid & 31;
  int gr = blockIdx.x * 8 + wave;
  if (gr >= H_ * N_) return;
  int h = gr / N_, n = gr % N_;
  const float *e1, *e2; int dd; float t;
  if (h < 2) {
    e1 = le1 + (h * N_ + n) * 32; e2 = le2 + h * 32 * N_; dd = 32;
    t = fminf(fmaxf(temp[h] * 2.0f, 0.1f), 5.0f);
  } else {
    int hg = h - 2;
    e1 = ge1 + (hg * N_ + n) * 64; e2 = ge2 + hg * 64 * N_; dd = 64;
    t = fminf(fmaxf(temp[h] * 0.5f, 0.1f), 2.0f);
  }
  float invt = 1.0f / t;
  float* s = sr[wave];
  for (int m = lane; m < N_; m += 32) {
    float acc = 0.0f;
    for (int d = 0; d < dd; ++d) acc += e1[d] * e2[d * N_ + m];
    s[m] = fmaxf(acc, 0.0f) * invt;
  }
  __builtin_amdgcn_wave_barrier();
  float mx = -3.4e38f;
  for (int j = lane; j < N_; j += 32) mx = fmaxf(mx, s[j]);
  mx = wred_max(mx);
  float sum = 0.0f;
  for (int j = lane; j < N_; j += 32) { float e = expf(s[j] - mx); s[j] = e; sum += e; }
  sum = wred_sum(sum);
  float inv = 1.0f / sum;
  for (int j = lane; j < N_; j += 32) s[j] *= inv;
  __builtin_amdgcn_wave_barrier();
  topk_renorm_write(s, N_, pv[wave], pi[wave], out + (size_t)(h * N_ + n) * N_, lane);
}

// ===========================================================================
// Kernel 2: fused temporal attention per (b,n).  WMMA-heavy, LDS-resident.
// ===========================================================================
// QKV projection GEMM: x(80x96 f16,row-major) @ wt(transposed weight) + bias.
template <bool TRANSPOSED_STORE>
__device__ __forceinline__ void proj_gemm(const _Float16* xh, const _Float16* wt,
                                          const float* bias_s, _Float16* dst,
                                          int wave, int lane) {
  for (int t = wave; t < 30; t += 8) {
    int mt = t / 6, nt = t % 6;
    v8f c = initC_bias(bias_s, nt * 16, lane);
#pragma unroll
    for (int kc = 0; kc < 3; ++kc) {
      v16h a = ldsA_v(xh, 96, mt * 16, kc * 32, 96, lane);
      v16h b = ldsBk_v(wt, 96, nt * 16, kc * 32, 96, lane);
      c = WMMA16(a, b, c);
    }
    int n  = nt * 16 + (lane & 15);
    int m0 = mt * 16 + ((lane & 16) ? 8 : 0);
    if (TRANSPOSED_STORE) {       // dst = vT[96][80]: one packed 16B store
      v8h pk;
#pragma unroll
      for (int r = 0; r < 8; ++r)
        pk[r] = (m0 < P_) ? (_Float16)c[r] : (_Float16)0.0f;   // run-uniform
      *(v8h*)(dst + n * 80 + m0) = pk;
    } else {                       // dst row-major [80][96]
#pragma unroll
      for (int r = 0; r < 8; ++r) {
        int m = m0 + r;
        dst[m * 96 + n] = (m < P_) ? (_Float16)c[r] : (_Float16)0.0f;
      }
    }
  }
}

__global__ void __launch_bounds__(256) attn_kernel(
    const float* __restrict__ patch, const float* __restrict__ pos,
    const float* __restrict__ wq, const float* __restrict__ bq,
    const float* __restrict__ wk, const float* __restrict__ bk,
    const float* __restrict__ wv, const float* __restrict__ bv,
    const float* __restrict__ wo, const float* __restrict__ bo,
    const float* __restrict__ tn_g, const float* __restrict__ tn_b,
    float* __restrict__ node, float* __restrict__ node_f) {
  // f16 region offsets (halves)
  constexpr int XH = 0;                  // x   [80][96] row-major (A)
  constexpr int WT = XH + 80 * 96;       // W^T [96(n)][96(k)]     (B,k-contig)
  constexpr int QH = WT + 96 * 96;       // q   [80][96] row-major (A for scores)
  constexpr int KH = QH + 80 * 96;       // k   [80][96] row-major (B,k-contig for scores)
  constexpr int VT = KH + 80 * 96;       // v^T [96(d)][80(p)]     (B,k-contig for o)
  constexpr int OH = VT + 96 * 80;       // o   [80][96] row-major (A for attended)
  constexpr int AH = OH + 80 * 96;       // att [80][96] row-major (A for o)
  constexpr int F16_TOT = AH + 80 * 96 + 32;  // +pad for speculative reads
  // f32 region offsets (floats)
  constexpr int BIASO = 0;
  constexpr int UNIO  = BIASO + 96;      // scores 80x80 / attended 80x96 (union)
  constexpr int PRMO  = UNIO + 80 * 96;
  constexpr int ROWMO = PRMO + 96;
  constexpr int REDO  = ROWMO + 80;
  constexpr int F32_TOT = REDO + 128;
  __shared__ __align__(16) unsigned char smem_raw[F16_TOT * 2 + F32_TOT * 4];
  _Float16* f16b = (_Float16*)smem_raw;
  float*    f32b = (float*)(smem_raw + F16_TOT * 2);
  _Float16* xh = f16b + XH; _Float16* wt = f16b + WT;
  _Float16* qh = f16b + QH; _Float16* kh = f16b + KH;
  _Float16* vT = f16b + VT; _Float16* oh = f16b + OH;
  _Float16* ah = f16b + AH;
  float* bias_s = f32b + BIASO; float* uni = f32b + UNIO;
  float* prm = f32b + PRMO; float* rowm = f32b + ROWMO; float* red = f32b + REDO;

  int tid = threadIdx.x, wave = tid >> 5, lane = tid & 31;
  int bn = blockIdx.x;
  const float* src = patch + (size_t)bn * (P_ * D_);

  for (int i = tid; i < 96; i += 256) prm[i] = 0.0f;
  for (int i = tid; i < 8 * 96; i += 256) xh[72 * 96 + i] = (_Float16)0.0f;
  __syncthreads();
  for (int i = tid; i < P_ * D_; i += 256) {
    float pvv = src[i];
    xh[i] = (_Float16)(pvv + pos[i]);
    atomicAdd(&prm[i % 96], pvv * (1.0f / 72.0f));
  }

  // ---- QKV projections (weights staged transposed: wt[n*96+k] = W[k][n]) ----
  __syncthreads();
  for (int i = tid; i < 96 * 96; i += 256) { int k = i / 96, n = i % 96; wt[n * 96 + k] = (_Float16)wq[i]; }
  for (int i = tid; i < 96; i += 256) bias_s[i] = bq[i];
  __syncthreads();
  proj_gemm<false>(xh, wt, bias_s, qh, wave, lane);
  __syncthreads();
  for (int i = tid; i < 96 * 96; i += 256) { int k = i / 96, n = i % 96; wt[n * 96 + k] = (_Float16)wk[i]; }
  for (int i = tid; i < 96; i += 256) bias_s[i] = bk[i];
  __syncthreads();
  proj_gemm<false>(xh, wt, bias_s, kh, wave, lane);
  __syncthreads();
  for (int i = tid; i < 96 * 96; i += 256) { int k = i / 96, n = i % 96; wt[n * 96 + k] = (_Float16)wv[i]; }
  for (int i = tid; i < 96; i += 256) bias_s[i] = bv[i];
  __syncthreads();
  proj_gemm<true>(xh, wt, bias_s, vT, wave, lane);
  __syncthreads();

  // ---- per-head attention ----
  float* s = uni;                               // 80x80 scores
  const float rscale = 0.20412414523193154f;    // 1/sqrt(24)
#pragma unroll 1
  for (int h = 0; h < NH_; ++h) {
    int co = h * HD_;
    for (int t = wave; t < 25; t += 8) {        // scores: q @ k^T
      int mt = t / 5, nt = t % 5;
      v8f c = {};
      v16h a = ldsA_v(qh, 96, mt * 16, co, co + HD_, lane);
      v16h b = ldsBk_v(kh, 96, nt * 16, co, co + HD_, lane);
      c = WMMA16(a, b, c);
      stC_f32(s, 80, mt * 16, nt * 16, c, rscale, lane);
    }
    __syncthreads();
    for (int r = wave; r < 80; r += 8) {        // row softmax -> att (f16)
      if (r < P_) {
        float mx = -3.4e38f;
        for (int j = lane; j < P_; j += 32) mx = fmaxf(mx, s[r * 80 + j]);
        mx = wred_max(mx);
        float sum = 0.0f;
        for (int j = lane; j < P_; j += 32) {
          float e = expf(s[r * 80 + j] - mx); s[r * 80 + j] = e; sum += e;
        }
        sum = wred_sum(sum);
        float inv = 1.0f / sum;
        for (int j = lane; j < 96; j += 32)
          ah[r * 96 + j] = (j < P_) ? (_Float16)(s[r * 80 + j] * inv) : (_Float16)0.0f;
      } else {
        for (int j = lane; j < 96; j += 32) ah[r * 96 + j] = (_Float16)0.0f;
      }
    }
    __syncthreads();
    for (int t = wave; t < 10; t += 8) {        // o_h = att @ v_h (v from vT)
      int mt = t / 2, nt = t % 2;
      v8f c = {};
#pragma unroll
      for (int kc = 0; kc < 3; ++kc) {
        v16h a = ldsA_v(ah, 96, mt * 16, kc * 32, 96, lane);
        v16h b = ldsBk_v(vT, 80, co + nt * 16, kc * 32, 80, lane);
        c = WMMA16(a, b, c);
      }
      int nl = nt * 16 + (lane & 15);
      int m0 = mt * 16 + ((lane & 16) ? 8 : 0);
      if (nl < HD_) {
        int col = co + nl;
#pragma unroll
        for (int r = 0; r < 8; ++r) {
          int m = m0 + r;
          oh[m * 96 + col] = (m < P_) ? (_Float16)c[r] : (_Float16)0.0f;
        }
      }
    }
    __syncthreads();
  }

  // ---- attended = o @ wo + bo ----
  for (int i = tid; i < 96 * 96; i += 256) { int k = i / 96, n = i % 96; wt[n * 96 + k] = (_Float16)wo[i]; }
  for (int i = tid; i < 96; i += 256) bias_s[i] = bo[i];
  __syncthreads();
  float* attd = uni;                            // 80x96 (reuses score region)
  for (int t = wave; t < 30; t += 8) {
    int mt = t / 6, nt = t % 6;
    v8f c = initC_bias(bias_s, nt * 16, lane);
#pragma unroll
    for (int kc = 0; kc < 3; ++kc) {
      v16h a = ldsA_v(oh, 96, mt * 16, kc * 32, 96, lane);
      v16h b = ldsBk_v(wt, 96, nt * 16, kc * 32, 96, lane);
      c = WMMA16(a, b, c);
    }
    stC_f32(attd, 96, mt * 16, nt * 16, c, 1.0f, lane);
  }
  __syncthreads();

  // ---- importance pooling + node + LN ----
  for (int r = wave; r < P_; r += 8) {
    float sm = 0.0f;
    for (int j = lane; j < 96; j += 32) sm += attd[r * 96 + j];
    sm = wred_sum(sm);
    if (lane == 0) rowm[r] = sm * (1.0f / 96.0f);
  }
  __syncthreads();
  if (wave == 0) {
    float mx = -3.4e38f;
    for (int r = lane; r < P_; r += 32) mx = fmaxf(mx, rowm[r]);
    mx = wred_max(mx);
    float sum = 0.0f;
    for (int r = lane; r < P_; r += 32) { float e = expf(rowm[r] - mx); rowm[r] = e; sum += e; }
    sum = wred_sum(sum);
    float inv = 1.0f / sum;
    for (int r = lane; r < P_; r += 32) rowm[r] *= inv;
  }
  __syncthreads();
  if (tid < 96) {
    float acc = 0.0f;
    for (int p2 = 0; p2 < P_; ++p2) acc += attd[p2 * 96 + tid] * rowm[p2];
    float nf = prm[tid];
    node_f[(size_t)bn * 96 + tid] = nf;
    red[tid] = acc + nf;
  }
  __syncthreads();
  if (wave == 0) {
    float v0 = red[lane], v1 = red[lane + 32], v2 = red[lane + 64];
    float s1 = wred_sum(v0 + v1 + v2);
    float s2 = wred_sum(v0 * v0 + v1 * v1 + v2 * v2);
    if (lane == 0) {
      float m = s1 * (1.0f / 96.0f);
      red[96] = m;
      red[97] = rsqrtf(s2 * (1.0f / 96.0f) - m * m + 1e-5f);
    }
  }
  __syncthreads();
  if (tid < 96)
    node[(size_t)bn * 96 + tid] = (red[tid] - red[96]) * red[97] * tn_g[tid] + tn_b[tid];
}

// ===========================================================================
// Kernel 3: dynamic encoder: node(96) -> ln/gelu(128) -> ln(64)
// ===========================================================================
__global__ void __launch_bounds__(256) dyn_enc_kernel(
    const float* __restrict__ node,
    const float* __restrict__ w1, const float* __restrict__ b1,
    const float* __restrict__ g1, const float* __restrict__ be1,
    const float* __restrict__ w2, const float* __restrict__ b2,
    const float* __restrict__ g2, const float* __restrict__ be2,
    float* __restrict__ dyn) {
  __shared__ float xrow[8][96];
  __shared__ float h1s[8][128];
  int tid = threadIdx.x, wave = tid >> 5, lane = tid & 31;
  int row = blockIdx.x * 8 + wave;
  for (int d = lane; d < 96; d += 32) xrow[wave][d] = node[(size_t)row * 96 + d];
  __builtin_amdgcn_wave_barrier();
  float v[4];
#pragma unroll
  for (int q = 0; q < 4; ++q) {
    int j = lane + 32 * q;
    float acc = b1[j];
    for (int d = 0; d < 96; ++d) acc += xrow[wave][d] * w1[d * 128 + j];
    v[q] = acc;
  }
  float s1 = wred_sum(v[0] + v[1] + v[2] + v[3]);
  float s2 = wred_sum(v[0] * v[0] + v[1] * v[1] + v[2] * v[2] + v[3] * v[3]);
  float mean = s1 * (1.0f / 128.0f);
  float inv = rsqrtf(s2 * (1.0f / 128.0f) - mean * mean + 1e-5f);
#pragma unroll
  for (int q = 0; q < 4; ++q) {
    int j = lane + 32 * q;
    h1s[wave][j] = gelu_exact((v[q] - mean) * inv * g1[j] + be1[j]);
  }
  __builtin_amdgcn_wave_barrier();
  float a0 = b2[lane], a1 = b2[lane + 32];
  for (int d = 0; d < 128; ++d) {
    float hv = h1s[wave][d];
    a0 += hv * w2[d * 64 + lane];
    a1 += hv * w2[d * 64 + lane + 32];
  }
  s1 = wred_sum(a0 + a1);
  s2 = wred_sum(a0 * a0 + a1 * a1);
  mean = s1 * (1.0f / 64.0f);
  inv = rsqrtf(s2 * (1.0f / 64.0f) - mean * mean + 1e-5f);
  dyn[(size_t)row * 64 + lane]      = (a0 - mean) * inv * g2[lane] + be2[lane];
  dyn[(size_t)row * 64 + lane + 32] = (a1 - mean) * inv * g2[lane + 32] + be2[lane + 32];
}

// ===========================================================================
// Kernel 4: one GNN layer.  dyn panel staged via TDM (fallback: plain copy).
// ===========================================================================
__global__ void __launch_bounds__(256) gnn_kernel(
    const float* __restrict__ dyn_in, const float* __restrict__ gw,
    const float* __restrict__ gb, const float* __restrict__ gg,
    const float* __restrict__ gbeta, float* __restrict__ dyn_out) {
  __shared__ float dl[N_ * 64];
  __shared__ float simr[8][360];
  __shared__ float aggs[8][64];
  int tid = threadIdx.x, wave = tid >> 5, lane = tid & 31;
  const float* db = dyn_in + (size_t)blockIdx.y * N_ * 64;
#ifdef USE_TDM
  if (wave == 0) {
    // 1-D TDM tile: 22912 fp32 elements (91,648 B), contiguous.
    unsigned long long ga = (unsigned long long)(size_t)db;
    unsigned ldso = (unsigned)(size_t)(float*)dl;     // LDS byte offset
    tdm_v4u g0;
    g0[0] = 1u;                                       // count=1 (valid D#)
    g0[1] = ldso;                                     // lds_addr
    g0[2] = (unsigned)(ga & 0xFFFFFFFFu);             // global_addr[31:0]
    g0[3] = (unsigned)((ga >> 32) & 0x01FFFFFFu) | 0x80000000u;  // addr[56:32]|type=2
    tdm_v8i g1;
    g1[0] = 0x00020000;                               // data_size=4B
    g1[1] = (int)(22912u << 16);                      // tensor_dim0[15:0]
    g1[2] = (int)((22912u >> 16) | (1u << 16));       // dim0[31:16] | tensor_dim1=1
    g1[3] = (int)(22912u << 16);                      // dim1 hi=0 | tile_dim0=22912
    g1[4] = 1;                                        // tile_dim1=1, tile_dim2=0
    g1[5] = 22912;                                    // tensor_dim0_stride lo
    g1[6] = (int)(22912u << 16);                      // stride0 hi=0 | stride1 lo
    g1[7] = 0;                                        // stride1 hi
    tdm_v4i g2 = {};
    tdm_v4i g3 = {};
    tdm_v8i g4 = {};
    __builtin_amdgcn_tensor_load_to_lds(g0, g1, g2, g3, g4, 0);
    __builtin_amdgcn_s_wait_tensorcnt(0);
  }
#else
  for (int i = tid; i < N_ * 64; i += 256) dl[i] = db[i];
#endif
  __syncthreads();
  int row = blockIdx.x * 8 + wave;
  if (row >= N_) return;
  float* sr = simr[wave];
  const float* rp = &dl[row * 64];
  float mx = -3.4e38f;
  for (int m = lane; m < N_; m += 32) {
    const float* mp = &dl[m * 64];
    float acc = 0.0f;
    for (int d = 0; d < 64; ++d) acc += rp[d] * mp[d];
    acc *= 5.0f;  // / 0.2
    sr[m] = acc;
    mx = fmaxf(mx, acc);
  }
  mx = wred_max(mx);
  float sum = 0.0f;
  for (int m = lane; m < N_; m += 32) { float e = expf(sr[m] - mx); sr[m] = e; sum += e; }
  sum = wred_sum(sum);
  float inv = 1.0f / sum;
  for (int m = lane; m < N_; m += 32) sr[m] *= inv;
  __builtin_amdgcn_wave_barrier();
  for (int dd = lane; dd < 64; dd += 32) {
    float acc = 0.0f;
    for (int m = 0; m < N_; ++m) acc += sr[m] * dl[m * 64 + dd];
    aggs[wave][dd] = acc;
  }
  __builtin_amdgcn_wave_barrier();
  float a0 = gb[lane], a1 = gb[lane + 32];
  for (int d = 0; d < 64; ++d) {
    float av = aggs[wave][d];
    a0 += av * gw[d * 64 + lane];
    a1 += av * gw[d * 64 + lane + 32];
  }
  float s1 = wred_sum(a0 + a1);
  float s2 = wred_sum(a0 * a0 + a1 * a1);
  float mean = s1 * (1.0f / 64.0f);
  float nrm = rsqrtf(s2 * (1.0f / 64.0f) - mean * mean + 1e-5f);
  float o0 = gelu_exact((a0 - mean) * nrm * gg[lane] + gbeta[lane]);
  float o1 = gelu_exact((a1 - mean) * nrm * gg[lane + 32] + gbeta[lane + 32]);
  size_t base = ((size_t)blockIdx.y * N_ + row) * 64;
  dyn_out[base + lane]      = o0 + rp[lane];
  dyn_out[base + lane + 32] = o1 + rp[lane + 32];
}

// ===========================================================================
// Kernel 5: dsim = relu(dyn @ static_e2)/t -> softmax -> topk.  WMMA GEMM.
// B panel staged TRANSPOSED so each lane's K run is contiguous (b128 loads).
// ===========================================================================
__global__ void __launch_bounds__(256) dyn_adj_kernel(
    const float* __restrict__ dyn, const float* __restrict__ se2,
    const float* __restrict__ temp, float* __restrict__ out) {
  __shared__ __align__(16) _Float16 aA[16 * 64];
  __shared__ __align__(16) _Float16 bBt[368 * 64];   // bBt[n*64+k] = e2[k][n]
  __shared__ float srow[16 * 368];
  __shared__ float pv[8][K_];
  __shared__ int   pi[8][K_];
  int tid = threadIdx.x, wave = tid >> 5, lane = tid & 31;
  int b = blockIdx.z, h = blockIdx.y, mt = blockIdx.x;
  for (int i = tid; i < 16 * 64; i += 256) {
    int r = mt * 16 + i / 64;
    aA[i] = (r < N_) ? (_Float16)dyn[((size_t)b * N_ + r) * 64 + (i % 64)] : (_Float16)0.0f;
  }
  const float* e2 = se2 + (size_t)h * 64 * N_;
  for (int i = tid; i < 64 * 368; i += 256) {
    int k = i / 368, n = i % 368;                     // coalesced global read
    bBt[n * 64 + k] = (n < N_) ? (_Float16)e2[k * N_ + n] : (_Float16)0.0f;
  }
  __syncthreads();
  float t = fminf(fmaxf(temp[h], 0.1f), 2.0f);
  float invt = 1.0f / t;
  for (int nt = wave; nt < 23; nt += 8) {
    v8f c = {};
#pragma unroll
    for (int kc = 0; kc < 2; ++kc) {
      v16h a  = ldsA_v(aA, 64, 0, kc * 32, 64, lane);
      v16h bt = ldsBk_v(bBt, 64, nt * 16, kc * 32, 64, lane);
      c = WMMA16(a, bt, c);
    }
    int n = nt * 16 + (lane & 15);
    int m0 = (lane & 16) ? 8 : 0;
#pragma unroll
    for (int r = 0; r < 8; ++r)
      srow[(m0 + r) * 368 + n] = fmaxf(c[r], 0.0f) * invt;
  }
  __syncthreads();
  for (int rr = wave; rr < 16; rr += 8) {
    int row = mt * 16 + rr;
    if (row >= N_) continue;
    float* sr = &srow[rr * 368];
    float mx = -3.4e38f;
    for (int j = lane; j < N_; j += 32) mx = fmaxf(mx, sr[j]);
    mx = wred_max(mx);
    float sum = 0.0f;
    for (int j = lane; j < N_; j += 32) { float e = expf(sr[j] - mx); sr[j] = e; sum += e; }
    sum = wred_sum(sum);
    float inv = 1.0f / sum;
    for (int j = lane; j < N_; j += 32) sr[j] *= inv;
    __builtin_amdgcn_wave_barrier();
    float* orow = out + (((size_t)b * H_ + h) * N_ + row) * N_;
    topk_renorm_write(sr, N_, pv[wave], pi[wave], orow, lane);
  }
}

// ===========================================================================
// Kernel 6: fusion gates fw = sigmoid(node_f @ gf_w + gf_b)
// ===========================================================================
__global__ void __launch_bounds__(256) fw_kernel(
    const float* __restrict__ node_f, const float* __restrict__ gf_w,
    const float* __restrict__ gf_b, float* __restrict__ fw) {
  int idx = blockIdx.x * 256 + threadIdx.x;
  if (idx >= BN_ * H_) return;
  int bn = idx / H_, h = idx % H_;
  float acc = gf_b[h];
  const float* nf = node_f + (size_t)bn * 96;
  for (int d = 0; d < 96; ++d) acc += nf[d] * gf_w[d * H_ + h];
  fw[idx] = sigmoidf_(acc);
}

// ===========================================================================
// Kernel 7: fusion + per-edge MLP + final.  One thread per edge.
// ===========================================================================
__global__ void __launch_bounds__(256) final_kernel(
    const float* __restrict__ sadj, const float* __restrict__ dadj,
    const float* __restrict__ fw,
    const float* __restrict__ w1, const float* __restrict__ b1,
    const float* __restrict__ g1, const float* __restrict__ be1,
    const float* __restrict__ w2, const float* __restrict__ b2,
    const float* __restrict__ w3, const float* __restrict__ b3,
    float* __restrict__ final_out) {
  size_t idx = (size_t)blockIdx.x * 256 + threadIdx.x;
  if (idx >= (size_t)B_ * NN_) return;
  int m = (int)(idx % N_);
  size_t t = idx / N_;
  int n = (int)(t % N_);
  int b = (int)(t / N_);
  float f[H_]; float meanh = 0.0f;
#pragma unroll
  for (int h = 0; h < H_; ++h) {
    float wv = fw[((size_t)b * N_ + n) * H_ + h];
    float sv = sadj[((size_t)h * N_ + n) * N_ + m];
    float dv = dadj[(((size_t)b * H_ + h) * N_ + n) * N_ + m];
    float fu = (1.0f - wv) * sv + wv * dv;
    f[h] = fu;
    meanh += fu * 0.25f;
  }
  float e1v[16];
  float ms = 0.0f, vs = 0.0f;
#pragma unroll
  for (int j = 0; j < 16; ++j) {
    float acc = b1[j];
#pragma unroll
    for (int h = 0; h < H_; ++h) acc += f[h] * w1[h * 16 + j];
    e1v[j] = acc;
    ms += acc;
  }
  ms *= (1.0f / 16.0f);
#pragma unroll
  for (int j = 0; j < 16; ++j) { float d = e1v[j] - ms; vs += d * d; }
  float inv = rsqrtf(vs * (1.0f / 16.0f) + 1e-5f);
#pragma unroll
  for (int j = 0; j < 16; ++j)
    e1v[j] = gelu_exact((e1v[j] - ms) * inv * g1[j] + be1[j]);
  float e2v[8];
#pragma unroll
  for (int j = 0; j < 8; ++j) {
    float acc = b2[j];
#pragma unroll
    for (int i = 0; i < 16; ++i) acc += e1v[i] * w2[i * 8 + j];
    e2v[j] = gelu_exact(acc);
  }
  float e3 = b3[0];
#pragma unroll
  for (int i = 0; i < 8; ++i) e3 += e2v[i] * w3[i];
  final_out[idx] = sigmoidf_(e3) * meanh;
}

// ===========================================================================
// Host launcher
// ===========================================================================
extern "C" void kernel_launch(void* const* d_in, const int* in_sizes, int n_in,
                              void* d_out, int out_size, void* d_ws, size_t ws_size,
                              hipStream_t stream) {
  (void)in_sizes; (void)n_in; (void)out_size; (void)ws_size;
  const float* patch = (const float*)d_in[0];
  const float* le1   = (const float*)d_in[1];
  const float* le2   = (const float*)d_in[2];
  const float* ge1   = (const float*)d_in[3];
  const float* ge2   = (const float*)d_in[4];
  const float* se2   = (const float*)d_in[5];
  const float* temp  = (const float*)d_in[6];
  const float* pos   = (const float*)d_in[7];
  const float* wq = (const float*)d_in[8];  const float* bq = (const float*)d_in[9];
  const float* wk = (const float*)d_in[10]; const float* bk = (const float*)d_in[11];
  const float* wv = (const float*)d_in[12]; const float* bv = (const float*)d_in[13];
  const float* wo = (const float*)d_in[14]; const float* bo = (const float*)d_in[15];
  const float* tn_g = (const float*)d_in[16]; const float* tn_b = (const float*)d_in[17];
  const float* de_w1 = (const float*)d_in[18]; const float* de_b1 = (const float*)d_in[19];
  const float* de_g1 = (const float*)d_in[20]; const float* de_be1 = (const float*)d_in[21];
  const float* de_w2 = (const float*)d_in[22]; const float* de_b2 = (const float*)d_in[23];
  const float* de_g2 = (const float*)d_in[24]; const float* de_be2 = (const float*)d_in[25];
  const float* gnn_w = (const float*)d_in[26]; const float* gnn_b = (const float*)d_in[27];
  const float* gnn_g = (const float*)d_in[28]; const float* gnn_be = (const float*)d_in[29];
  const float* gf_w = (const float*)d_in[30]; const float* gf_b = (const float*)d_in[31];
  const float* ee_w1 = (const float*)d_in[32]; const float* ee_b1 = (const float*)d_in[33];
  const float* ee_g  = (const float*)d_in[34]; const float* ee_be = (const float*)d_in[35];
  const float* ee_w2 = (const float*)d_in[36]; const float* ee_b2 = (const float*)d_in[37];
  const float* ee_w3 = (const float*)d_in[38]; const float* ee_b3 = (const float*)d_in[39];

  float* out = (float*)d_out;
  float* out_final  = out;
  float* out_static = out + (size_t)B_ * NN_;
  float* out_dyn    = out_static + (size_t)H_ * NN_;

  float* ws = (float*)d_ws;
  float* node   = ws;
  float* node_f = node + (size_t)BN_ * 96;
  float* dynA   = node_f + (size_t)BN_ * 96;
  float* dynB   = dynA + (size_t)BN_ * 64;
  float* fwbuf  = dynB + (size_t)BN_ * 64;

  static_adj_kernel<<<179, 256, 0, stream>>>(le1, le2, ge1, ge2, temp, out_static);
  attn_kernel<<<BN_, 256, 0, stream>>>(patch, pos, wq, bq, wk, bk, wv, bv, wo, bo,
                                       tn_g, tn_b, node, node_f);
  dyn_enc_kernel<<<716, 256, 0, stream>>>(node, de_w1, de_b1, de_g1, de_be1,
                                          de_w2, de_b2, de_g2, de_be2, dynA);
  gnn_kernel<<<dim3(45, 16), 256, 0, stream>>>(dynA, gnn_w, gnn_b, gnn_g, gnn_be, dynB);
  gnn_kernel<<<dim3(45, 16), 256, 0, stream>>>(dynB, gnn_w + 64 * 64, gnn_b + 64,
                                               gnn_g + 64, gnn_be + 64, dynA);
  dyn_adj_kernel<<<dim3(23, 4, 16), 256, 0, stream>>>(dynA, se2, temp, out_dyn);
  fw_kernel<<<(BN_ * H_ + 255) / 256, 256, 0, stream>>>(node_f, gf_w, gf_b, fwbuf);
  final_kernel<<<(int)(((size_t)B_ * NN_ + 255) / 256), 256, 0, stream>>>(
      out_static, out_dyn, fwbuf, ee_w1, ee_b1, ee_g, ee_be, ee_w2, ee_b2, ee_w3, ee_b3,
      out_final);
}
